// AttentionBlock_23227183137082
// MI455X (gfx1250) — matrix operand order repaired and checked by reading, not compiled, not run on hardware
//
#include <hip/hip_runtime.h>
#include <hip/hip_bf16.h>
#include <stdint.h>

// ---------------------------------------------------------------------------
// Problem constants: x [B=2, H=64, W=64, C=512], 32 groups, single-head attn
// ---------------------------------------------------------------------------
#define BATCH   2
#define N_PIX   4096      // H*W
#define C_DIM   512
#define GROUPS  32
#define CG      16        // C/GROUPS
#define MTOT    (BATCH * N_PIX)   // 8192 rows

typedef __attribute__((ext_vector_type(16))) __bf16 v16bf;
typedef __attribute__((ext_vector_type(8)))  float  v8f;
typedef __attribute__((ext_vector_type(4)))  unsigned int u32x4;
typedef __attribute__((ext_vector_type(8)))  int  i32x8;
typedef __attribute__((ext_vector_type(4)))  int  i32x4;

// float -> bf16, round-to-nearest-even
__device__ __forceinline__ __bf16 f2bf(float f) {
  unsigned int u = __builtin_bit_cast(unsigned int, f);
  u += 0x7FFFu + ((u >> 16) & 1u);
  unsigned short h = (unsigned short)(u >> 16);
  return __builtin_bit_cast(__bf16, h);
}

// ---------------------------------------------------------------------------
// Tensor Data Mover: async 2-D tile load Global->LDS (CDNA5 ISA ch. 7/8).
// D# group0: count=1, lds_addr, 57b global_addr, type=2 ("image").
// D# group1: data_size=2B, tensor dims == tile dims, row stride in elements,
// optional LDS row padding (pad_enable/pad_interval/pad_amount) to break
// bank conflicts. Groups 2/3 = 0 (2-D). Tracked by TENSORcnt.
// padbits: 0, or (1<<20)|(interval_code<<22)|(amount_code<<25).
// ---------------------------------------------------------------------------
#define TDM_PAD_32DW_4DW  ((1u << 20) | (4u << 22) | (3u << 25))  // 128B rows -> +16B
#define TDM_PAD_64DW_4DW  ((1u << 20) | (5u << 22) | (3u << 25))  // 256B rows -> +16B

__device__ __forceinline__ void tdm_load_2d(unsigned lds_off, const void* gptr,
                                            int tile_w, int tile_h,
                                            int stride_elems, unsigned padbits) {
  const unsigned long long ga = (unsigned long long)(size_t)gptr;
  u32x4 g0;
  g0[0] = 1u;                                            // count=1, user mode
  g0[1] = lds_off;                                       // LDS byte address
  g0[2] = (unsigned)(ga & 0xffffffffull);                // global_addr[31:0]
  g0[3] = (unsigned)((ga >> 32) & 0x01ffffffull) | 0x80000000u;  // type=2
  i32x8 g1;
  g1[0] = (int)(0x00010000u | padbits);   // wg_mask=0, data_size=2B, padding
  g1[1] = (tile_w & 0xffff) << 16;        // tensor_dim0[15:0]
  g1[2] = ((unsigned)tile_w >> 16) | ((tile_h & 0xffff) << 16);  // d0 hi, d1 lo
  g1[3] = (tile_w & 0xffff) << 16;        // tensor_dim1 hi=0, tile_dim0
  g1[4] = tile_h & 0xffff;                // tile_dim1 (tile_dim2=0)
  g1[5] = stride_elems;                   // tensor_dim0_stride[31:0]
  g1[6] = 0;
  g1[7] = 0;
  const i32x4 z4 = {0, 0, 0, 0};
  const i32x8 z8 = {0, 0, 0, 0, 0, 0, 0, 0};
  __builtin_amdgcn_tensor_load_to_lds(g0, g1, z4, z4, z8, 0);
}

__device__ __forceinline__ unsigned lds_off_of(const void* p) {
  return (unsigned)(size_t)p;
}

// ---------------------------------------------------------------------------
// WMMA fragment gathers (wave32, 16x16x32 bf16; layouts per CDNA5 ISA 7.12.2)
// A-frag: lane row = lane&15, K = {hf*8+0..7, 16+hf*8+0..7} (2x16B runs).
// ---------------------------------------------------------------------------
__device__ __forceinline__ v16bf load_afrag(const __bf16* src, int ld, int r0,
                                            int c0, int lane) {
  const int hf = (lane >> 4) & 1;
  const __bf16* p = src + (size_t)(r0 + (lane & 15)) * ld + c0 + hf * 8;
  v16bf a;
#pragma unroll
  for (int e = 0; e < 16; ++e) a[e] = p[((e >> 3) << 4) + (e & 7)];
  return a;
}

// B-frag (32x16, KxN) where the operand is stored K-contiguous per output
// column: src is row-major [N][K] (i.e. the transposed operand). Per-lane
// K = {hf*8+0..7, 16+hf*8+0..7} (2x16B runs), matching the A-frag.
__device__ __forceinline__ v16bf load_btfrag(const __bf16* src, int ld, int k0,
                                             int n0, int lane) {
  const int hf = (lane >> 4) & 1;
  const __bf16* p = src + (size_t)(n0 + (lane & 15)) * ld + k0 + hf * 8;
  v16bf b;
#pragma unroll
  for (int e = 0; e < 16; ++e) b[e] = p[((e >> 3) << 4) + (e & 7)];
  return b;
}

__device__ __forceinline__ v8f wmma_bf16(v16bf a, v16bf b, v8f c) {
  return __builtin_amdgcn_wmma_f32_16x16x32_bf16(false, a, false, b, (short)0,
                                                 c, false, false);
}

// ---------------------------------------------------------------------------
// Kernel 1: GroupNorm statistics. One block per (batch, group).
// ---------------------------------------------------------------------------
__global__ void __launch_bounds__(256) gn_stats(const float* __restrict__ x,
                                                float* __restrict__ mean,
                                                float* __restrict__ rstd) {
  __shared__ float ssum[256], ssq[256];
  const int bg = blockIdx.x;           // 0..63
  const int b = bg >> 5, g = bg & 31;
  const float* base = x + (size_t)b * N_PIX * C_DIM + g * CG;
  float s = 0.f, q = 0.f;
  for (int i = threadIdx.x; i < N_PIX * CG; i += 256) {
    const int hw = i >> 4, cc = i & 15;
    const float v = base[(size_t)hw * C_DIM + cc];
    s += v;
    q += v * v;
  }
  ssum[threadIdx.x] = s;
  ssq[threadIdx.x] = q;
  __syncthreads();
  for (int st = 128; st > 0; st >>= 1) {
    if (threadIdx.x < st) {
      ssum[threadIdx.x] += ssum[threadIdx.x + st];
      ssq[threadIdx.x] += ssq[threadIdx.x + st];
    }
    __syncthreads();
  }
  if (threadIdx.x == 0) {
    const float inv = 1.f / (float)(N_PIX * CG);
    const float m = ssum[0] * inv;
    const float var = ssq[0] * inv - m * m;
    mean[bg] = m;
    rstd[bg] = rsqrtf(var + 1e-5f);
  }
}

// ---------------------------------------------------------------------------
// Kernel 2: normalize + affine, write bf16 h.
// ---------------------------------------------------------------------------
__global__ void __launch_bounds__(256) gn_apply(const float* __restrict__ x,
                                                const float* __restrict__ gamma,
                                                const float* __restrict__ beta,
                                                const float* __restrict__ mean,
                                                const float* __restrict__ rstd,
                                                __bf16* __restrict__ h) {
  const size_t idx = (size_t)blockIdx.x * 256 + threadIdx.x;
  const int c = (int)(idx & (C_DIM - 1));
  const size_t n = idx >> 9;
  const int b = (int)(n >> 12);
  const int bg = b * GROUPS + (c >> 4);
  const float v = (x[idx] - mean[bg]) * rstd[bg] * gamma[c] + beta[c];
  h[idx] = f2bf(v);
}

// ---------------------------------------------------------------------------
// Kernel 3: cast + TRANSPOSE the four CxC weight matrices to bf16.
// wT[d][c] = w[c][d] so GEMM B-fragments are K-contiguous per lane.
// ---------------------------------------------------------------------------
__global__ void __launch_bounds__(256) cast_w(const float* __restrict__ wq,
                                              const float* __restrict__ wk,
                                              const float* __restrict__ wv,
                                              const float* __restrict__ wp,
                                              __bf16* __restrict__ wqt,
                                              __bf16* __restrict__ wkt,
                                              __bf16* __restrict__ wvt,
                                              __bf16* __restrict__ wpt) {
  const size_t i = (size_t)blockIdx.x * 256 + threadIdx.x;  // < 512*512
  const int c = (int)(i >> 9), d = (int)(i & 511);
  const size_t o = (size_t)d * C_DIM + c;
  wqt[o] = f2bf(wq[i]);
  wkt[o] = f2bf(wk[i]);
  wvt[o] = f2bf(wv[i]);
  wpt[o] = f2bf(wp[i]);
}

// ---------------------------------------------------------------------------
// Kernel 4/6: tiled WMMA GEMM: D[M x 512] = (A[M x 512] * Wt^T + bias) * mul
// Wt is pre-transposed [d][c]. 64x64 tile / 256-thread block, TDM staging
// with LDS row padding (pitch 72 elems). TRANSOUT stores D transposed
// ([col][row], pitch MTOT) -- used for V so flash P*V B-frags are contiguous.
// ---------------------------------------------------------------------------
#define GLD 72    // padded LDS pitch for 64-elem rows

template <bool PROJ, bool TRANSOUT>
__global__ void __launch_bounds__(256) gemm512(const __bf16* __restrict__ A,
                                               const __bf16* __restrict__ Wt,
                                               const float* __restrict__ bias,
                                               const float* __restrict__ resid,
                                               __bf16* __restrict__ outb,
                                               float* __restrict__ outf,
                                               float mul) {
  __shared__ __bf16 As[64 * GLD];
  __shared__ __bf16 Ws[64 * GLD];
  const int tid = threadIdx.x, lane = tid & 31, w = tid >> 5;
  const int hf = (lane >> 4) & 1, l15 = lane & 15;
  const int row0 = blockIdx.x * 64, col0 = blockIdx.y * 64;
  const int t0 = w * 2;
  const int mi = (t0 >> 2) * 16;          // same row-block for both tiles
  const int ni0 = (t0 & 3) * 16, ni1 = ni0 + 16;
  const unsigned As_off = lds_off_of(As) + w * 8 * GLD * 2;
  const unsigned Ws_off = lds_off_of(Ws) + w * 8 * GLD * 2;
  const v8f vzero = {0.f, 0.f, 0.f, 0.f, 0.f, 0.f, 0.f, 0.f};
  v8f acc0 = vzero, acc1 = vzero;

  for (int kc = 0; kc < C_DIM; kc += 64) {
    __syncthreads();
    // TDM: each wave DMAs its 8-row slice of A rows and Wt rows (cols d)
    tdm_load_2d(As_off, A + (size_t)(row0 + w * 8) * C_DIM + kc, 64, 8, C_DIM,
                TDM_PAD_32DW_4DW);
    tdm_load_2d(Ws_off, Wt + (size_t)(col0 + w * 8) * C_DIM + kc, 64, 8, C_DIM,
                TDM_PAD_32DW_4DW);
    __builtin_amdgcn_s_wait_tensorcnt(0);
    __syncthreads();
#pragma unroll
    for (int kk = 0; kk < 64; kk += 32) {
      const v16bf af = load_afrag(As, GLD, mi, kk, lane);
      const v16bf b0 = load_btfrag(Ws, GLD, kk, ni0, lane);
      const v16bf b1 = load_btfrag(Ws, GLD, kk, ni1, lane);
      acc0 = wmma_bf16(af, b0, acc0);
      acc1 = wmma_bf16(af, b1, acc1);
    }
  }
  // epilogue
#pragma unroll
  for (int t = 0; t < 2; ++t) {
    const v8f acc = t ? acc1 : acc0;
    const int col = col0 + (t ? ni1 : ni0) + l15;
#pragma unroll
    for (int v = 0; v < 8; ++v) {
      const int row = row0 + mi + v + 8 * hf;
      const float val = (acc[v] + bias[col]) * mul;
      if (PROJ) {
        const size_t o = (size_t)row * C_DIM + col;
        outf[o] = val + resid[o];
      } else if (TRANSOUT) {
        outb[(size_t)col * MTOT + row] = f2bf(val);
      } else {
        outb[(size_t)row * C_DIM + col] = f2bf(val);
      }
    }
  }
}

// ---------------------------------------------------------------------------
// Kernel 5: flash attention. One block = 64 queries, stream 64-key blocks.
// Wave w owns O column blocks {w, w+8, w+16, w+24}; all 8 waves compute in
// every P*V channel chunk. V arrives TRANSPOSED ([ch][pixel]) so its
// B-fragments are contiguous. Q is pre-scaled by 1/sqrt(C).
// LDS pitches padded (136 / 72 / 65) to spread banks.
// ---------------------------------------------------------------------------
#define QKLD 136   // padded pitch for 128-elem rows
#define VLD  72    // padded pitch for 64-elem rows
#define SLD  65
#define PLD  72

__global__ void __launch_bounds__(256) flash_attn(const __bf16* __restrict__ Q,
                                                  const __bf16* __restrict__ K,
                                                  const __bf16* __restrict__ Vt,
                                                  __bf16* __restrict__ O) {
  __shared__ __bf16 Qc[64 * QKLD];     // Q chunk [64 q][128 ch]     17408 B
  __shared__ __bf16 KVs[128 * VLD];    // K [64][136] / Vt [128][72] 18432 B
  __shared__ float Ss[64 * SLD];       // scores                     16640 B
  __shared__ __bf16 Ps[64 * PLD];      // probabilities               9216 B
  __shared__ float red[256];           // softmax 4-way reduction     1024 B
  __shared__ float mrow[64], lrow[64], arow[64];

  const int tid = threadIdx.x, lane = tid & 31, w = tid >> 5;
  const int hf = (lane >> 4) & 1, l15 = lane & 15;
  const int blk = blockIdx.x;       // 0..127
  const int b = blk >> 6;
  const int q0 = (blk & 63) * 64;
  const __bf16* Qb = Q + ((size_t)b * N_PIX + q0) * C_DIM;
  const __bf16* Kb = K + (size_t)b * N_PIX * C_DIM;
  const unsigned Qc_off = lds_off_of(Qc) + w * 8 * QKLD * 2;
  const unsigned Kc_off = lds_off_of(KVs) + w * 8 * QKLD * 2;
  const unsigned Vc_off = lds_off_of(KVs) + w * 16 * VLD * 2;

  const v8f vzero = {0.f, 0.f, 0.f, 0.f, 0.f, 0.f, 0.f, 0.f};
  v8f accO[16];                     // [rb][chunk] ; col = chunk*128 + w*16
#pragma unroll
  for (int i = 0; i < 16; ++i) accO[i] = vzero;
  if (tid < 64) { mrow[tid] = -1e30f; lrow[tid] = 0.f; }

  const int sm = (w * 2 >> 2) * 16;           // S-tile row block for this wave
  const int sn0 = ((w * 2) & 3) * 16, sn1 = sn0 + 16;
  const int r = tid & 63, seg = tid >> 6;     // softmax row / 16-col segment

  for (int kb = 0; kb < 64; ++kb) {
    const __bf16* Kt = Kb + (size_t)kb * 64 * C_DIM;
    v8f s0 = vzero, s1 = vzero;

    // ---- scores: S = Q * K^T over 4 channel chunks of 128 (TDM staged) ----
    for (int c4 = 0; c4 < 4; ++c4) {
      __syncthreads();
      tdm_load_2d(Qc_off, Qb + (size_t)(w * 8) * C_DIM + c4 * 128, 128, 8,
                  C_DIM, TDM_PAD_64DW_4DW);
      tdm_load_2d(Kc_off, Kt + (size_t)(w * 8) * C_DIM + c4 * 128, 128, 8,
                  C_DIM, TDM_PAD_64DW_4DW);
      __builtin_amdgcn_s_wait_tensorcnt(0);
      __syncthreads();
#pragma unroll
      for (int kk = 0; kk < 128; kk += 32) {
        const v16bf af = load_afrag(Qc, QKLD, sm, kk, lane);
        const v16bf b0 = load_btfrag(KVs, QKLD, kk, sn0, lane);
        const v16bf b1 = load_btfrag(KVs, QKLD, kk, sn1, lane);
        s0 = wmma_bf16(af, b0, s0);
        s1 = wmma_bf16(af, b1, s1);
      }
    }
#pragma unroll
    for (int v = 0; v < 8; ++v) {
      const int rr = sm + v + 8 * hf;
      Ss[rr * SLD + sn0 + l15] = s0[v];
      Ss[rr * SLD + sn1 + l15] = s1[v];
    }
    __syncthreads();

    // ---- online softmax: 4 threads per row, 16 cols each ----
    {
      float mx = -1e30f;
#pragma unroll
      for (int j = 0; j < 16; ++j)
        mx = fmaxf(mx, Ss[r * SLD + seg * 16 + j]);
      red[seg * 64 + r] = mx;
      __syncthreads();
      const float mold = mrow[r];
      float mnew = fmaxf(fmaxf(red[r], red[64 + r]),
                         fmaxf(red[128 + r], red[192 + r]));
      mnew = fmaxf(mnew, mold);
      __syncthreads();
      float sum = 0.f;
#pragma unroll
      for (int j = 0; j < 16; ++j) {
        const float pv = __expf(Ss[r * SLD + seg * 16 + j] - mnew);
        sum += pv;
        Ps[r * PLD + seg * 16 + j] = f2bf(pv);
      }
      red[seg * 64 + r] = sum;
      __syncthreads();
      if (seg == 0) {
        const float al = __expf(mold - mnew);
        arow[r] = al;
        mrow[r] = mnew;
        lrow[r] = lrow[r] * al +
                  (red[r] + red[64 + r] + red[128 + r] + red[192 + r]);
      }
      __syncthreads();
    }

    // ---- rescale O accumulators by alpha ----
#pragma unroll
    for (int rb = 0; rb < 4; ++rb) {
#pragma unroll
      for (int v = 0; v < 8; ++v) {
        const float al = arow[rb * 16 + v + 8 * hf];
#pragma unroll
        for (int cb = 0; cb < 4; ++cb) accO[rb * 4 + cb][v] *= al;
      }
    }

    // ---- O += P * V over 4 channel chunks; all waves active each chunk ----
    // Vt chunk: rows = channels c4*128..+128, cols = keys kb*64..+64
    for (int c4 = 0; c4 < 4; ++c4) {
      __syncthreads();
      tdm_load_2d(Vc_off,
                  Vt + (size_t)(c4 * 128 + w * 16) * MTOT + b * N_PIX + kb * 64,
                  64, 16, MTOT, TDM_PAD_32DW_4DW);
      __builtin_amdgcn_s_wait_tensorcnt(0);
      __syncthreads();
#pragma unroll
      for (int kk = 0; kk < 64; kk += 32) {
        // B[k=key][n=ch] = Vt_lds[ch][key]: contiguous per lane
        const v16bf bf = load_btfrag(KVs, VLD, kk, w * 16, lane);
#pragma unroll
        for (int rb = 0; rb < 4; ++rb) {
          const v16bf af = load_afrag(Ps, PLD, rb * 16, kk, lane);
          accO[rb * 4 + c4] = wmma_bf16(af, bf, accO[rb * 4 + c4]);
        }
      }
    }
  }

  // ---- finalize: divide by l, store bf16; col = chunk*128 + w*16 ----
#pragma unroll
  for (int rb = 0; rb < 4; ++rb) {
#pragma unroll
    for (int j = 0; j < 4; ++j) {
      const int col = j * 128 + w * 16 + l15;
#pragma unroll
      for (int v = 0; v < 8; ++v) {
        const int row = rb * 16 + v + 8 * hf;
        const float val = accO[rb * 4 + j][v] / lrow[row];
        O[((size_t)b * N_PIX + q0 + row) * C_DIM + col] = f2bf(val);
      }
    }
  }
}

// ---------------------------------------------------------------------------
// Host launcher
// ---------------------------------------------------------------------------
extern "C" void kernel_launch(void* const* d_in, const int* in_sizes, int n_in,
                              void* d_out, int out_size, void* d_ws,
                              size_t ws_size, hipStream_t stream) {
  const float* x     = (const float*)d_in[0];
  const float* gamma = (const float*)d_in[1];
  const float* beta  = (const float*)d_in[2];
  const float* wq    = (const float*)d_in[3];
  const float* bq    = (const float*)d_in[4];
  const float* wk    = (const float*)d_in[5];
  const float* bk    = (const float*)d_in[6];
  const float* wv    = (const float*)d_in[7];
  const float* bv    = (const float*)d_in[8];
  const float* wp    = (const float*)d_in[9];
  const float* bp    = (const float*)d_in[10];
  float* out = (float*)d_out;

  // carve workspace (256B aligned)
  char* p = (char*)d_ws;
  auto carve = [&](size_t bytes) {
    void* r = (void*)p;
    p += (bytes + 255) & ~(size_t)255;
    return r;
  };
  const size_t nelem = (size_t)MTOT * C_DIM;  // 4M
  float*  mean = (float*)carve(64 * sizeof(float));
  float*  rstd = (float*)carve(64 * sizeof(float));
  __bf16* h    = (__bf16*)carve(nelem * 2);
  __bf16* qb   = (__bf16*)carve(nelem * 2);
  __bf16* kb   = (__bf16*)carve(nelem * 2);
  __bf16* vtb  = (__bf16*)carve(nelem * 2);   // V stored transposed [ch][pix]
  __bf16* ob   = (__bf16*)carve(nelem * 2);
  __bf16* wqt  = (__bf16*)carve((size_t)C_DIM * C_DIM * 2);
  __bf16* wkt  = (__bf16*)carve((size_t)C_DIM * C_DIM * 2);
  __bf16* wvt  = (__bf16*)carve((size_t)C_DIM * C_DIM * 2);
  __bf16* wpt  = (__bf16*)carve((size_t)C_DIM * C_DIM * 2);

  const float scale = 0.044194173824159216f;  // 1/sqrt(512), folded into Q

  // 1) GroupNorm stats
  gn_stats<<<dim3(64), 256, 0, stream>>>(x, mean, rstd);
  // 2) normalize + affine -> bf16
  gn_apply<<<dim3((unsigned)(nelem / 256)), 256, 0, stream>>>(x, gamma, beta,
                                                              mean, rstd, h);
  // 3) weights -> bf16, transposed
  cast_w<<<dim3(1024), 256, 0, stream>>>(wq, wk, wv, wp, wqt, wkt, wvt, wpt);
  // 4) Q/K/V projections (Q pre-scaled by 1/sqrt(C); V output transposed)
  const dim3 ggrid(MTOT / 64, C_DIM / 64);
  gemm512<false, false><<<ggrid, 256, 0, stream>>>(h, wqt, bq, nullptr, qb,
                                                   nullptr, scale);
  gemm512<false, false><<<ggrid, 256, 0, stream>>>(h, wkt, bk, nullptr, kb,
                                                   nullptr, 1.0f);
  gemm512<false, true><<<ggrid, 256, 0, stream>>>(h, wvt, bv, nullptr, vtb,
                                                  nullptr, 1.0f);
  // 5) flash attention
  flash_attn<<<dim3(MTOT / 64), 256, 0, stream>>>(qb, kb, vtb, ob);
  // 6) output projection + residual -> f32 d_out
  gemm512<true, false><<<ggrid, 256, 0, stream>>>(ob, wpt, bp, x, nullptr, out,
                                                  1.0f);
}
